// WordStyleEncoder_44899588113030
// MI455X (gfx1250) — matrix operand order repaired and checked by reading, not compiled, not run on hardware
//
#include <hip/hip_runtime.h>

// ---------------------------------------------------------------------------
// WordStyleEncoder for MI455X (gfx1250, wave32, WMMA).
//   B=64, T=1024, IDIM=80, H=256, 2 GRU layers, 4-head attention over 5 tokens
// Roofline: ~103 GFLOP total, ~52 GFLOP inside 2048 sequential GRU steps;
// all activations/weights fit in the 192MB L2 -> recurrence-latency bound.
// Design: batch-parallel GRU recurrence (4 blocks x 16 rows, no grid sync),
// bf16 WMMA (v_wmma_f32_16x16x32_bf16) f32-accumulate, f32 hidden state in
// LDS, weights pre-swizzled into B-fragment order, double-buffered staging
// via gfx1250 async-to-LDS (ASYNCcnt); GEMM staging split load/store so the
// global-load wait is covered by the WMMAs.
// ---------------------------------------------------------------------------

typedef __bf16        v16bf __attribute__((ext_vector_type(16)));
typedef __bf16        v4bf  __attribute__((ext_vector_type(4)));
typedef float         v8f   __attribute__((ext_vector_type(8)));
typedef unsigned int  v8u   __attribute__((ext_vector_type(8)));
typedef int           v4i   __attribute__((ext_vector_type(4)));

#if __has_builtin(__builtin_amdgcn_global_load_async_to_lds_b128) && \
    __has_builtin(__builtin_amdgcn_s_wait_asynccnt)
#define HAVE_ASYNC_LDS 1
#else
#define HAVE_ASYNC_LDS 0
#endif

#define GLOBAL_AS __attribute__((address_space(1)))
#define LDS_AS    __attribute__((address_space(3)))

__device__ __forceinline__ float fast_sigmoid(float x) {
  return 1.0f / (1.0f + __expf(-x));
}
__device__ __forceinline__ float fast_tanh(float x) {
  return 1.0f - 2.0f / (__expf(2.0f * x) + 1.0f);
}

// ---------------------------------------------------------------------------
// Swizzle an NxK f32 weight matrix into WMMA B-fragment order (bf16, K padded
// to KC*32): out[((nt*KC+kc)*32+lane)*16+e] = W[nt*16+(lane&15)][kc*32+(lane>>4)*8+(e&7)+(e>>3)*16]
// ---------------------------------------------------------------------------
__global__ __launch_bounds__(256) void swizzle_w(const float* __restrict__ W,
                                                 __bf16* __restrict__ out,
                                                 int N, int K, int KC) {
  int idx = blockIdx.x * 256 + threadIdx.x;
  int total = (N / 16) * KC * 512;
  if (idx >= total) return;
  int e    = idx & 15;
  int lane = (idx >> 4) & 31;
  int rest = idx >> 9;
  int kc   = rest % KC;
  int nt   = rest / KC;
  int n = nt * 16 + (lane & 15);
  int k = kc * 32 + (lane >> 4) * 8 + (e & 7) + (e >> 3) * 16;
  float v = (k < K) ? W[(size_t)n * K + k] : 0.0f;
  out[idx] = (__bf16)v;
}

// ---------------------------------------------------------------------------
// k/v projection of the 5 style tokens (tiny: 5x256 each).
// ---------------------------------------------------------------------------
__global__ __launch_bounds__(256) void kv_proj(const float* __restrict__ st,
                                               const float* __restrict__ kw,
                                               const float* __restrict__ vw,
                                               const float* __restrict__ in_b,
                                               float* __restrict__ kbuf,
                                               float* __restrict__ vbuf) {
  int idx = blockIdx.x * 256 + threadIdx.x;
  if (idx >= 5 * 256) return;
  int s = idx / 256, c = idx % 256;
  float aK = 0.f, aV = 0.f;
  for (int i = 0; i < 128; ++i) {
    float x = st[s * 128 + i];
    aK += x * kw[c * 128 + i];
    aV += x * vw[c * 128 + i];
  }
  kbuf[idx] = aK + in_b[256 + c];
  vbuf[idx] = aV + in_b[512 + c];
}

// ---------------------------------------------------------------------------
// Generic WMMA GEMM: Out[M x N] = (A[M x Kreal] @ Wswz^T + bias) * scale
// Block = 256 threads (8 waves), tile 64x64. A tile double-buffered in LDS;
// staging is split: global float4 loads issued right after the barrier,
// bf16 convert + ds_store AFTER the WMMAs (load latency hidden by matrix ops).
// ---------------------------------------------------------------------------
__global__ __launch_bounds__(256) void gemm_bias_wmma(
    const float*  __restrict__ A,
    const __bf16* __restrict__ Wswz,
    const float*  __restrict__ bias,
    void*         __restrict__ Out,
    int M, int Kreal, int KC, int N, float scale, int storeBf16) {
  __shared__ __align__(16) __bf16 As[2][64][48];  // stride 48 bf16 (96B rows)

  const int tid  = threadIdx.x;
  const int lane = tid & 31;
  const int w    = tid >> 5;      // 8 waves
  const int mi   = w & 3;         // m sub-tile 0..3
  const int ni0  = (w >> 2) * 2;  // this wave's pair of n sub-tiles
  const int m0   = blockIdx.x * 64;
  const int n0   = blockIdx.y * 64;

  float4 pend[2];  // in-flight staged A values (2 x float4 per thread)

  auto stage_load = [&](int kc) {
    bool full = (kc * 32 + 32 <= Kreal);  // uniform
#pragma unroll
    for (int g = 0; g < 2; ++g) {
      int u   = tid + g * 256;  // 0..511
      int row = u >> 3;
      int kkb = (u & 7) * 4;
      int k   = kc * 32 + kkb;
      const float* ap = A + (size_t)(m0 + row) * Kreal;
      if (full) {
        pend[g] = *(const float4*)(ap + k);
      } else {  // tail chunk: clamped addresses, branchless zero select
        float4 v4;
        v4.x = ap[(k + 0) < Kreal ? (k + 0) : 0]; v4.x = (k + 0) < Kreal ? v4.x : 0.f;
        v4.y = ap[(k + 1) < Kreal ? (k + 1) : 0]; v4.y = (k + 1) < Kreal ? v4.y : 0.f;
        v4.z = ap[(k + 2) < Kreal ? (k + 2) : 0]; v4.z = (k + 2) < Kreal ? v4.z : 0.f;
        v4.w = ap[(k + 3) < Kreal ? (k + 3) : 0]; v4.w = (k + 3) < Kreal ? v4.w : 0.f;
        pend[g] = v4;
      }
    }
  };
  auto stage_store = [&](int p) {
#pragma unroll
    for (int g = 0; g < 2; ++g) {
      int u   = tid + g * 256;
      int row = u >> 3;
      int kkb = (u & 7) * 4;
      v4bf pk;
      pk[0] = (__bf16)pend[g].x; pk[1] = (__bf16)pend[g].y;
      pk[2] = (__bf16)pend[g].z; pk[3] = (__bf16)pend[g].w;
      *(v4bf*)&As[p][row][kkb] = pk;
    }
  };

  v8f acc0 = {}; v8f acc1 = {};
  const v16bf* Wv = (const v16bf*)Wswz;
  const int nt0 = blockIdx.y * 4 + ni0;

  stage_load(0);
  stage_store(0);
  for (int kc = 0; kc < KC; ++kc) {
    const int p = kc & 1;
    __syncthreads();                          // As[p] ready; As[1-p] free
    if (kc + 1 < KC) stage_load(kc + 1);      // issue loads, no wait yet

    // A fragment (16x32 bf16 ISA layout) from As[p]
    v8u raw;
    {
      int mrow = mi * 16 + (lane & 15);
      int kb   = (lane >> 4) * 8;
      const unsigned int* arow = (const unsigned int*)&As[p][mrow][0];
#pragma unroll
      for (int q = 0; q < 4; ++q) {
        raw[q]     = arow[(kb >> 1) + q];
        raw[q + 4] = arow[((kb + 16) >> 1) + q];
      }
    }
    v16bf afrag = __builtin_bit_cast(v16bf, raw);

    v16bf b0 = Wv[((size_t)nt0 * KC + kc) * 32 + lane];
    v16bf b1 = Wv[((size_t)(nt0 + 1) * KC + kc) * 32 + lane];
    acc0 = __builtin_amdgcn_wmma_f32_16x16x32_bf16(false, afrag, false, b0,
                                                   (short)0, acc0, false, false);
    acc1 = __builtin_amdgcn_wmma_f32_16x16x32_bf16(false, afrag, false, b1,
                                                   (short)0, acc1, false, false);

    if (kc + 1 < KC) stage_store(1 - p);      // wait lands after the WMMAs
  }

  // store: C/D layout -> lane: n = lane&15 ; elem e: m = e + 8*(lane>>4)
  const int mBase = mi * 16 + (lane >> 4) * 8;
#pragma unroll
  for (int half = 0; half < 2; ++half) {
    v8f acc  = half ? acc1 : acc0;
    int col  = n0 + (ni0 + half) * 16 + (lane & 15);
    float bc = bias[col];
#pragma unroll
    for (int e = 0; e < 8; ++e) {
      int row = m0 + mBase + e;
      float v = (acc[e] + bc) * scale;
      if (storeBf16) ((__bf16*)Out)[(size_t)row * N + col] = (__bf16)v;
      else           ((float*) Out)[(size_t)row * N + col] = v;
    }
  }
}

// ---------------------------------------------------------------------------
// GRU recurrence. Grid = 4 blocks x 16 batch rows, no inter-block sync. 512
// threads = 16 waves; wave w owns hidden columns w*16..w*16+15 (gate tiles
// nt = w, 16+w, 32+w; K=256 -> 8 bf16 WMMAs per tile). h kept f32 in LDS
// with a bf16 shadow as the WMMA A operand. The (16x768 bf16) xg slice for
// step t+1 is staged into a double buffer during step t's WMMAs — via the
// gfx1250 async-to-LDS path (ASYNCcnt) when available.
// ---------------------------------------------------------------------------
__global__ __launch_bounds__(512) void gru_recur(
    const __bf16* __restrict__ xg,      // (B, T, 768) bf16
    const __bf16* __restrict__ WhhSwz,  // 48 nt x 8 kc x 32 lanes x 16 bf16
    const float*  __restrict__ bhh,     // 768
    float*        __restrict__ outs,    // (B, T, 256) f32
    int T) {
  __shared__ float  hF[16][260];                   // f32 hidden state
  __shared__ __align__(16) __bf16 hB[16][264];     // bf16 shadow (WMMA A)
  __shared__ __align__(16) __bf16 xgs[2][16][776]; // double-buffered xg slice

  const int tid  = threadIdx.x;
  const int lane = tid & 31;
  const int w    = tid >> 5;  // 0..15
  const int b0   = blockIdx.x * 16;

  // zero-init hidden state
  {
    float* hf = &hF[0][0];
    for (int i = tid; i < 16 * 260; i += 512) hf[i] = 0.0f;
    __bf16* hb = &hB[0][0];
    for (int i = tid; i < 16 * 264; i += 512) hb[i] = (__bf16)0.0f;
  }

  const int   j  = w * 16 + (lane & 15);  // owned hidden column
  const float br = bhh[j];
  const float bz = bhh[256 + j];
  const float bn = bhh[512 + j];
  const int   mBase = (lane >> 4) * 8;

  // stage timestep-t xg slice into buffer p (wave w copies its row: 1536B)
  auto stage = [&](int t, int p) {
#if HAVE_ASYNC_LDS
    char* src = (char*)xg + ((size_t)(b0 + w) * T + t) * 1536;
    char* dst = (char*)&xgs[p][w][0];
#pragma unroll
    for (int i = 0; i < 3; ++i) {  // 3 x (32 lanes x 16B) = 1536B
      int boff = (lane + i * 32) * 16;
      __builtin_amdgcn_global_load_async_to_lds_b128(
          (GLOBAL_AS v4i*)(src + boff), (LDS_AS v4i*)(dst + boff), 0, 0);
    }
#else
    const unsigned int* src =
        (const unsigned int*)xg + ((size_t)(b0 + w) * T + t) * 384;
    unsigned int* dst = (unsigned int*)&xgs[p][w][0];
#pragma unroll
    for (int r = 0; r < 12; ++r) dst[lane + r * 32] = src[lane + r * 32];
#endif
  };

  stage(0, 0);
  __syncthreads();  // h zero-init visible (xg slice 0 handled by waits below)

  const v16bf* Wv = (const v16bf*)WhhSwz;
  const int ntR = w, ntZ = 16 + w, ntN = 32 + w;

  for (int t = 0; t < T; ++t) {
    const int p = t & 1;

    // A fragments (16x32 bf16) from the bf16 hidden shadow
    v16bf afrag[8];
#pragma unroll
    for (int kc = 0; kc < 8; ++kc) {
      v8u raw;
      int m  = lane & 15;
      int kb = kc * 32 + (lane >> 4) * 8;
      const unsigned int* hrow = (const unsigned int*)&hB[m][0];
#pragma unroll
      for (int q = 0; q < 4; ++q) {
        raw[q]     = hrow[(kb >> 1) + q];
        raw[q + 4] = hrow[((kb + 16) >> 1) + q];
      }
      afrag[kc] = __builtin_bit_cast(v16bf, raw);
    }
#if HAVE_ASYNC_LDS
    __builtin_amdgcn_s_wait_asynccnt(0);  // xgs[p] DMA (issued last step) done
#endif
    __syncthreads();  // xgs[p] visible to all; all waves done reading hB

    if (t + 1 < T) stage(t + 1, 1 - p);  // overlaps the 24 WMMAs below

    // gh tiles for this wave's columns: r, z, n
    v8f accR = {}, accZ = {}, accN = {};
#pragma unroll
    for (int kc = 0; kc < 8; ++kc) {
      v16bf bR = Wv[((size_t)ntR * 8 + kc) * 32 + lane];
      v16bf bZ = Wv[((size_t)ntZ * 8 + kc) * 32 + lane];
      v16bf bN = Wv[((size_t)ntN * 8 + kc) * 32 + lane];
      accR = __builtin_amdgcn_wmma_f32_16x16x32_bf16(false, afrag[kc], false, bR,
                                                     (short)0, accR, false, false);
      accZ = __builtin_amdgcn_wmma_f32_16x16x32_bf16(false, afrag[kc], false, bZ,
                                                     (short)0, accZ, false, false);
      accN = __builtin_amdgcn_wmma_f32_16x16x32_bf16(false, afrag[kc], false, bN,
                                                     (short)0, accN, false, false);
    }

    // gate math in f32; update h and emit output
#pragma unroll
    for (int e = 0; e < 8; ++e) {
      int m = mBase + e;
      float xr = (float)xgs[p][m][j];
      float xz = (float)xgs[p][m][256 + j];
      float xn = (float)xgs[p][m][512 + j];
      float gr = accR[e] + br;
      float gz = accZ[e] + bz;
      float gn = accN[e] + bn;
      float r  = fast_sigmoid(xr + gr);
      float z  = fast_sigmoid(xz + gz);
      float n  = fast_tanh(xn + r * gn);
      float hOld = hF[m][j];
      float hNew = (1.0f - z) * n + z * hOld;
      hF[m][j] = hNew;
      hB[m][j] = (__bf16)hNew;
      outs[((size_t)(b0 + m) * T + t) * 256 + j] = hNew;
    }
    __syncthreads();  // h fully updated before next step's A-frag reads
  }
}

// ---------------------------------------------------------------------------
// Attention context: one thread per (row, head). q already scaled by 1/8.
// ---------------------------------------------------------------------------
__global__ __launch_bounds__(256) void attn_ctx(const float* __restrict__ q,
                                                const float* __restrict__ kbuf,
                                                const float* __restrict__ vbuf,
                                                float* __restrict__ ctx) {
  int idx  = blockIdx.x * 256 + threadIdx.x;  // row*4 + head
  int head = idx & 3;
  int row  = idx >> 2;
  const float* qp = q + (size_t)row * 256 + head * 64;
  float qr[64];
#pragma unroll
  for (int i = 0; i < 64; ++i) qr[i] = qp[i];

  float s[5];
  float mx = -1e30f;
#pragma unroll
  for (int t = 0; t < 5; ++t) {
    const float* kp = kbuf + t * 256 + head * 64;
    float d = 0.0f;
#pragma unroll
    for (int i = 0; i < 64; ++i) d += qr[i] * kp[i];
    s[t] = d;
    mx   = fmaxf(mx, d);
  }
  float den = 0.0f;
#pragma unroll
  for (int t = 0; t < 5; ++t) { s[t] = __expf(s[t] - mx); den += s[t]; }
  float inv = 1.0f / den;

  float o[64];
#pragma unroll
  for (int i = 0; i < 64; ++i) o[i] = 0.0f;
#pragma unroll
  for (int t = 0; t < 5; ++t) {
    const float* vp = vbuf + t * 256 + head * 64;
    float pw = s[t] * inv;
#pragma unroll
    for (int i = 0; i < 64; ++i) o[i] += pw * vp[i];
  }
  float* op = ctx + (size_t)row * 256 + head * 64;
#pragma unroll
  for (int i = 0; i < 64; ++i) op[i] = o[i];
}

// ---------------------------------------------------------------------------
// Host orchestration. ws peak ~226 MiB (xg bf16 96 MiB + 2x 64 MiB f32 +
// ~1.5 MiB swizzled weights); xg region reused for q, outs0 reused for ctx.
// ---------------------------------------------------------------------------
extern "C" void kernel_launch(void* const* d_in, const int* in_sizes, int n_in,
                              void* d_out, int out_size, void* d_ws, size_t ws_size,
                              hipStream_t stream) {
  (void)in_sizes; (void)n_in; (void)out_size; (void)ws_size;
  const float* feats  = (const float*)d_in[0];
  const float* styleT = (const float*)d_in[1];
  const float* Wih0   = (const float*)d_in[2];
  const float* Whh0   = (const float*)d_in[3];
  const float* bih0   = (const float*)d_in[4];
  const float* bhh0   = (const float*)d_in[5];
  const float* Wih1   = (const float*)d_in[6];
  const float* Whh1   = (const float*)d_in[7];
  const float* bih1   = (const float*)d_in[8];
  const float* bhh1   = (const float*)d_in[9];
  const float* q_w    = (const float*)d_in[10];
  const float* k_w    = (const float*)d_in[11];
  const float* v_w    = (const float*)d_in[12];
  const float* in_b   = (const float*)d_in[13];
  const float* out_w  = (const float*)d_in[14];
  const float* out_b  = (const float*)d_in[15];

  char* ws = (char*)d_ws;
  size_t off = 0;
  auto alloc = [&](size_t bytes) -> void* {
    void* p = ws + off;
    off = (off + bytes + 255) & ~(size_t)255;
    return p;
  };

  const int M = 64 * 1024;  // B*T rows

  __bf16* wih0s = (__bf16*)alloc((size_t)48 * 3 * 512 * 2);
  __bf16* whh0s = (__bf16*)alloc((size_t)48 * 8 * 512 * 2);
  __bf16* wih1s = (__bf16*)alloc((size_t)48 * 8 * 512 * 2);
  __bf16* whh1s = (__bf16*)alloc((size_t)48 * 8 * 512 * 2);
  __bf16* qws   = (__bf16*)alloc((size_t)16 * 8 * 512 * 2);
  __bf16* ows   = (__bf16*)alloc((size_t)16 * 8 * 512 * 2);
  float*  kbuf  = (float*)alloc(5 * 256 * 4);
  float*  vbuf  = (float*)alloc(5 * 256 * 4);
  __bf16* xg    = (__bf16*)alloc((size_t)M * 768 * 2);  // reused as q (f32)
  float*  outs0 = (float*)alloc((size_t)M * 256 * 4);   // reused as ctx
  float*  outs1 = (float*)alloc((size_t)M * 256 * 4);
  float*  qbuf  = (float*)xg;
  float*  ctx   = outs0;

  swizzle_w<<<288, 256, 0, stream>>>(Wih0, wih0s, 768, 80, 3);
  swizzle_w<<<768, 256, 0, stream>>>(Whh0, whh0s, 768, 256, 8);
  swizzle_w<<<768, 256, 0, stream>>>(Wih1, wih1s, 768, 256, 8);
  swizzle_w<<<768, 256, 0, stream>>>(Whh1, whh1s, 768, 256, 8);
  swizzle_w<<<256, 256, 0, stream>>>(q_w, qws, 256, 256, 8);
  swizzle_w<<<256, 256, 0, stream>>>(out_w, ows, 256, 256, 8);
  kv_proj<<<5, 256, 0, stream>>>(styleT, k_w, v_w, in_b, kbuf, vbuf);

  gemm_bias_wmma<<<dim3(M / 64, 12), 256, 0, stream>>>(
      feats, wih0s, bih0, xg, M, 80, 3, 768, 1.0f, 1);
  gru_recur<<<4, 512, 0, stream>>>(xg, whh0s, bhh0, outs0, 1024);

  gemm_bias_wmma<<<dim3(M / 64, 12), 256, 0, stream>>>(
      outs0, wih1s, bih1, xg, M, 256, 8, 768, 1.0f, 1);
  gru_recur<<<4, 512, 0, stream>>>(xg, whh1s, bhh1, outs1, 1024);

  gemm_bias_wmma<<<dim3(M / 64, 4), 256, 0, stream>>>(
      outs1, qws, in_b, qbuf, M, 256, 8, 256, 0.125f, 0);

  attn_ctx<<<1024, 256, 0, stream>>>(qbuf, kbuf, vbuf, ctx);

  gemm_bias_wmma<<<dim3(M / 64, 4), 256, 0, stream>>>(
      ctx, ows, out_b, (float*)d_out, M, 256, 8, 256, 1.0f, 0);
}
